// Attention_b_44925357916244
// MI455X (gfx1250) — matrix operand order repaired and checked by reading, not compiled, not run on hardware
//
#include <hip/hip_runtime.h>
#include <cstdint>
#include <cstddef>

// Problem constants (from the reference).
constexpr int kB = 32;
constexpr int kS = 2048;
constexpr int kH = 1024;
constexpr int kA = 256;

typedef __attribute__((ext_vector_type(16))) __bf16 v16bf;
typedef __attribute__((ext_vector_type(8)))  float  v8f;

// Pack two f32 -> packed bf16 pair (low = a, high = b).
// Round-half-up (+0x8000) then pack the high halves with a single v_perm_b32.
__device__ __forceinline__ unsigned pk2bf(float a, float b) {
  union { float f; unsigned u; } xa, xb;
  xa.f = a; xb.f = b;
  const unsigned ra = xa.u + 0x8000u;
  const unsigned rb = xb.u + 0x8000u;
  // sel 0x07060302: result bytes {rb[3],rb[2],ra[3],ra[2]} = {bf(b),bf(a)}
  return __builtin_amdgcn_perm(rb, ra, 0x07060302u);
}

__device__ __forceinline__ float fast_rcp(float x) {
#if __has_builtin(__builtin_amdgcn_rcpf)
  return __builtin_amdgcn_rcpf(x);
#else
  return 1.0f / x;
#endif
}

__device__ __forceinline__ float tanh_fast(float x) {
  // tanh(x) = 1 - 2/(1+exp(2x)); saturates correctly via exp overflow/underflow.
  return 1.0f - 2.0f * fast_rcp(1.0f + __expf(2.0f * x));
}

// ---------------------------------------------------------------------------
// Kernel 0: one-shot W1 -> bf16 copy (row-major [A][H]) so the GEMM's B
// fragments are raw uint4 loads with no per-iteration conversion.
// ---------------------------------------------------------------------------
__global__ __launch_bounds__(256) void w1_to_bf16_kernel(
    const float* __restrict__ W, unsigned short* __restrict__ Wbf) {
  const int idx = blockIdx.x * 256 + threadIdx.x;  // 8 elements per thread
  const int e   = idx * 8;                         // kH divisible by 8: no row cross
  const int a   = e >> 10;                         // / kH
  const int col = e & (kH - 1);
  const float4* src = (const float4*)(W + (size_t)a * (2 * kH) + col);
  float4 f0 = src[0], f1 = src[1];
  uint4 o;
  o.x = pk2bf(f0.x, f0.y); o.y = pk2bf(f0.z, f0.w);
  o.z = pk2bf(f1.x, f1.y); o.w = pk2bf(f1.z, f1.w);
  *(uint4*)(Wbf + e) = o;
}

// ---------------------------------------------------------------------------
// Kernel 1: c[b,a] = bias[a] + sum_h W[a, H+h] * h_t[b,h]   (tiny)
// ---------------------------------------------------------------------------
__global__ __launch_bounds__(256) void precompute_c_kernel(
    const float* __restrict__ W, const float* __restrict__ h_t,
    const float* __restrict__ bias, float* __restrict__ c_ws) {
  const int bb = blockIdx.x;      // batch
  const int a  = threadIdx.x;     // attention unit
  const float4* wrow = (const float4*)(W + (size_t)a * (2 * kH) + kH);
  const float4* hrow = (const float4*)(h_t + (size_t)bb * kH);
  float acc = bias[a];
#pragma unroll 4
  for (int i = 0; i < kH / 4; ++i) {
    float4 w = wrow[i]; float4 h = hrow[i];
    acc += w.x * h.x + w.y * h.y + w.z * h.z + w.w * h.w;
  }
  c_ws[bb * kA + a] = acc;
}

// ---------------------------------------------------------------------------
// Kernel 2: fused  y[b,s] = u . tanh( h_i[b] @ W1^T + c[b] )
// BF16 WMMA GEMM: per block, M-tile = 64 rows of S, N = full A = 256,
// K = H = 1024 in steps of 32. 16 waves: wave w owns columns [16w,16w+16).
// Double-buffered LDS A-tile, one barrier per K-step, prefetched global loads,
// pre-converted bf16 W1 (B fragments load with zero VALU).
// ---------------------------------------------------------------------------
constexpr int MT   = 64;   // S rows per block
constexpr int KT   = 32;   // K step (bf16 WMMA depth)
constexpr int LDSS = 40;   // LDS row stride in bf16 elems (80B pad: conflict-free b128)
constexpr int ATSZ = MT * LDSS;

union AFrag { v16bf v; uint4 q[2]; };

__global__ __launch_bounds__(512) void attn_gemm_kernel(
    const float* __restrict__ h_i, const unsigned short* __restrict__ Wbf,
    const float* __restrict__ c_ws, const float* __restrict__ u,
    float* __restrict__ y_ws) {
  const int b    = blockIdx.y;
  const int s0   = blockIdx.x * MT;
  const int t    = threadIdx.x;
  const int wave = t >> 5;
  const int lane = t & 31;
  const int half = lane >> 4;   // 0: lanes 0-15, 1: lanes 16-31
  const int ln   = lane & 15;
  const int a0   = wave * 16;   // this wave's N-strip in A

  __shared__ alignas(16) unsigned short aT[2 * ATSZ];  // double-buffered 64x32 bf16 tile
  __shared__ float ypart[16][MT];                      // per-wave row partials

  v8f acc[4] = {};  // 4 M-subtiles x (16x16 f32)

  const int    srow  = t >> 3;                          // 0..63 : staged row
  const int    sc4   = t & 7;                           // float4 col within K-slab
  const int    stoff = srow * LDSS + sc4 * 4;           // LDS store offset (elems)
  const float* hrow  = h_i + ((size_t)b * kS + s0 + srow) * kH + sc4 * 4;
  const unsigned short* wrow = Wbf + (size_t)(a0 + ln) * kH + half * 16;

  // Prologue: stage K-slab 0 into buffer 0.
  {
    float4 g = *(const float4*)(hrow);
    uint2 pk; pk.x = pk2bf(g.x, g.y); pk.y = pk2bf(g.z, g.w);
    *(uint2*)&aT[stoff] = pk;
  }
  __syncthreads();

  constexpr int NIT = kH / KT;
  for (int it = 0; it < NIT; ++it) {
    const int k0 = it * KT;
    const unsigned short* cur = aT + (it & 1) * ATSZ;
    unsigned short*       nxt = aT + ((it + 1) & 1) * ATSZ;

    // Issue all global loads for this iteration up front:
    //  - B fragment: 16 contiguous bf16 in contiguous k order, same as the A fragment
    //    (column a0+ln; lanes 0-15: K 0-15, lanes 16-31: K 16-31; VGPR v = K{2v,2v+1})
    //  - prefetch of the next A K-slab
    AFrag bf;
    bf.q[0] = *(const uint4*)(wrow + k0);
    bf.q[1] = *(const uint4*)(wrow + k0 + 8);
    const bool more = (it + 1 < NIT);
    float4 gn;
    if (more) gn = *(const float4*)(hrow + k0 + KT);

    // Load all 4 A fragments (8x ds_load_b128), then 4 WMMAs.
    // A fragment in contiguous k order (same as B):
    //   lanes 0-15: K 0-15 ; lanes 16-31: K 16-31 ; row = ln.
    AFrag af[4];
#pragma unroll
    for (int mt = 0; mt < 4; ++mt) {
      const int r = mt * 16 + ln;
      af[mt].q[0] = *(const uint4*)&cur[r * LDSS + half * 16];       // K {0|16}
      af[mt].q[1] = *(const uint4*)&cur[r * LDSS + half * 16 + 8];   // K {8|24}
    }
#pragma unroll
    for (int mt = 0; mt < 4; ++mt) {
      acc[mt] = __builtin_amdgcn_wmma_f32_16x16x32_bf16(
          false, af[mt].v, false, bf.v, (short)0, acc[mt], false, false);
    }

    // Stage the prefetched slab into the other buffer.
    if (more) {
      uint2 pk; pk.x = pk2bf(gn.x, gn.y); pk.y = pk2bf(gn.z, gn.w);
      *(uint2*)&nxt[stoff] = pk;
    }
    __syncthreads();  // single barrier: publishes nxt, retires reads of cur
  }

  // Epilogue: tanh(acc + c) * u, reduce over this wave's 16 columns.
  const float cc = c_ws[b * kA + a0 + ln];
  const float uu = u[a0 + ln];
#pragma unroll
  for (int mt = 0; mt < 4; ++mt) {
#pragma unroll
    for (int j = 0; j < 8; ++j) {
      float p = tanh_fast(acc[mt][j] + cc) * uu;
#pragma unroll
      for (int off = 8; off >= 1; off >>= 1) p += __shfl_xor(p, off, 16);
      if (ln == 0) ypart[wave][mt * 16 + half * 8 + j] = p;  // row of C/D tile
    }
  }
  __syncthreads();
  if (t < MT) {   // deterministic cross-wave sum in fixed order
    float sum = 0.0f;
#pragma unroll
    for (int w = 0; w < 16; ++w) sum += ypart[w][t];
    y_ws[(size_t)b * kS + s0 + t] = sum;
  }
}

// ---------------------------------------------------------------------------
// Kernel 3: masked-scatter (cross-batch cumsum gather) + softmax over S.
// One block per batch row.
// ---------------------------------------------------------------------------
__global__ __launch_bounds__(256) void softmax_kernel(
    const float* __restrict__ y_ws, const unsigned char* __restrict__ mask,
    float* __restrict__ alpha_ws) {
  const int b = blockIdx.x;
  const int t = threadIdx.x;
  __shared__ float red[256];

  float beta[8];
  float mx = -3.4e38f;
#pragma unroll
  for (int i = 0; i < 8; ++i) {
    const int s = i * 256 + t;
    int cnt = 0;
    for (int bb = 0; bb <= b; ++bb) cnt += (int)mask[(size_t)bb * kS + s];
    int pos = cnt - 1; if (pos < 0) pos = 0;
    float v = mask[(size_t)b * kS + s] ? y_ws[(size_t)pos * kS + s] : -1e20f;
    beta[i] = v;
    mx = fmaxf(mx, v);
  }
  red[t] = mx; __syncthreads();
  for (int o = 128; o > 0; o >>= 1) { if (t < o) red[t] = fmaxf(red[t], red[t + o]); __syncthreads(); }
  mx = red[0]; __syncthreads();

  float e[8]; float sum = 0.0f;
#pragma unroll
  for (int i = 0; i < 8; ++i) { e[i] = __expf(beta[i] - mx); sum += e[i]; }
  red[t] = sum; __syncthreads();
  for (int o = 128; o > 0; o >>= 1) { if (t < o) red[t] += red[t + o]; __syncthreads(); }
  const float inv = fast_rcp(red[0]);
#pragma unroll
  for (int i = 0; i < 8; ++i) alpha_ws[(size_t)b * kS + i * 256 + t] = e[i] * inv;
}

// ---------------------------------------------------------------------------
// Kernel 4: out[b,h] = sum_s alpha[b,s] * h_i[b,s,h]   (bandwidth-bound)
// ---------------------------------------------------------------------------
__global__ __launch_bounds__(256) void wsum_kernel(
    const float* __restrict__ h_i, const float* __restrict__ alpha_ws,
    float* __restrict__ out) {
  const int b = blockIdx.y;
  const int h = blockIdx.x * 256 + threadIdx.x;
  __shared__ float al[kS];
  for (int i = threadIdx.x; i < kS; i += 256) al[i] = alpha_ws[(size_t)b * kS + i];
  __syncthreads();
  const float* base = h_i + (size_t)b * kS * kH + h;
  float acc = 0.0f;
#pragma unroll 8
  for (int s = 0; s < kS; ++s) acc += al[s] * base[(size_t)s * kH];
  out[(size_t)b * kH + h] = acc;
}

// ---------------------------------------------------------------------------
// Host launcher.
// Workspace: Wbf (512KB) | c (32KB) | y (256KB) | alpha (256KB) = ~1.06MB.
// ---------------------------------------------------------------------------
extern "C" void kernel_launch(void* const* d_in, const int* in_sizes, int n_in,
                              void* d_out, int out_size, void* d_ws, size_t ws_size,
                              hipStream_t stream) {
  (void)in_sizes; (void)n_in; (void)out_size; (void)ws_size;
  const float*         h_i  = (const float*)d_in[0];
  const float*         h_t  = (const float*)d_in[1];
  const unsigned char* mask = (const unsigned char*)d_in[2];  // jax bool = 1 byte
  const float*         W    = (const float*)d_in[3];
  const float*         bias = (const float*)d_in[4];
  const float*         u    = (const float*)d_in[5];
  float*               out  = (float*)d_out;

  char*           ws       = (char*)d_ws;
  unsigned short* Wbf      = (unsigned short*)(ws);
  float*          c_ws     = (float*)(ws + 512 * 1024);
  float*          y_ws     = (float*)(ws + 512 * 1024 + 32 * 1024);
  float*          alpha_ws = (float*)(ws + 512 * 1024 + 32 * 1024 + 256 * 1024);

  w1_to_bf16_kernel<<<dim3((kA * kH / 8) / 256), dim3(256), 0, stream>>>(W, Wbf);
  precompute_c_kernel<<<dim3(kB), dim3(kA), 0, stream>>>(W, h_t, bias, c_ws);
  attn_gemm_kernel<<<dim3(kS / MT, kB), dim3(512), 0, stream>>>(h_i, Wbf, c_ws, u, y_ws);
  softmax_kernel<<<dim3(kB), dim3(256), 0, stream>>>(y_ws, mask, alpha_ws);
  wsum_kernel<<<dim3(kH / 256, kB), dim3(256), 0, stream>>>(h_i, alpha_ws, out);
}
